// PaperGNN_70815420776874
// MI455X (gfx1250) — matrix operand order repaired and checked by reading, not compiled, not run on hardware
//
#include <hip/hip_runtime.h>
#include <hip/hip_bf16.h>

// ---------------------------------------------------------------------------
// PaperGNN for MI455X (gfx1250, wave32, WMMA).
// All GEMMs run as bf16 WMMA (v_wmma_f32_16x16x32_bf16) with fp32 accumulate.
// Block = 128 threads = 4 waves; each wave computes a 16x64 output tile.
// Weights are pre-packed into per-lane fragment order so a B-fragment is two
// coalesced global b128 loads; A-fragments are two ds_read_b128 per lane.
// ---------------------------------------------------------------------------

typedef __attribute__((ext_vector_type(16))) __bf16 v16bf;
typedef __attribute__((ext_vector_type(8)))  float  v8f;

union BFrag { v16bf v; uint4 q[2]; unsigned short s[16]; };

__device__ __forceinline__ unsigned short f2bf(float f) {
    unsigned int u = __float_as_uint(f);
    u += 0x7FFFu + ((u >> 16) & 1u);        // round-to-nearest-even
    return (unsigned short)(u >> 16);
}
__device__ __forceinline__ float bf2f(unsigned short h) {
    return __uint_as_float(((unsigned int)h) << 16);
}

// ---------------------------------------------------------------------------
// WMMA fragment layouts (ISA 7.12.2, wave32):
//   A 16x32 bf16 : M = lane&15 ; lane<16: elems = K 0..7,16..23
//                                lane>=16: elems = K 8..15,24..31
//     -> per lane: two contiguous 8-element (16B) runs => 2x ds_read_b128
//   B 32x16 bf16 : N = lane&15 ; lane<16: K 0..7,16..23 ; lane>=16: K 8..15,24..31
//     -> packed buffer: lane's 16 elems contiguous  => 2x global_load_b128
//   C/D 16x16 f32: col = lane&15 ; row = r + (lane&16 ? 8 : 0)
// ---------------------------------------------------------------------------
template <int K, int LDA>
__device__ __forceinline__ void wmma_tile(const unsigned short* __restrict__ A,
                                          const unsigned short* __restrict__ Wp,
                                          v8f acc[4]) {
    const int lane = threadIdx.x & 31;
    const int m    = lane & 15;
    const int kbA  = (lane & 16) ? 8 : 0;
#pragma unroll
    for (int kc = 0; kc < (K >> 5); ++kc) {
        BFrag af;
        const uint4* ap = (const uint4*)(A + m * LDA + kc * 32 + kbA);
        af.q[0] = ap[0];                 // K = kc*32+kbA   .. +7
        af.q[1] = ap[2];                 // K = kc*32+kbA+16.. +23
        const uint4* bp = (const uint4*)(Wp + ((size_t)(kc * 4) * 32 + lane) * 16);
#pragma unroll
        for (int nb = 0; nb < 4; ++nb) {
            BFrag bf;
            bf.q[0] = bp[nb * 64];       // each nb: +32 lanes * 16 elems = 64 uint4
            bf.q[1] = bp[nb * 64 + 1];
            acc[nb] = __builtin_amdgcn_wmma_f32_16x16x32_bf16(
                false, af.v, false, bf.v, (short)0, acc[nb], false, false);
        }
    }
}

// bias + ReLU + bf16 store (row-major, stride ldo); fast path for full tiles
__device__ __forceinline__ void epilogue_relu(const v8f acc[4], const float* __restrict__ bias,
                                              unsigned short* __restrict__ out, int ldo,
                                              int rows_valid) {
    const int lane  = threadIdx.x & 31;
    const int rbase = (lane & 16) ? 8 : 0;
    const int n0    = lane & 15;
    if (rows_valid >= 16) {
#pragma unroll
        for (int nb = 0; nb < 4; ++nb) {
            const int col = nb * 16 + n0;
            const float b = bias[col];
#pragma unroll
            for (int r = 0; r < 8; ++r) {
                float v = acc[nb][r] + b;
                out[(rbase + r) * ldo + col] = f2bf(v > 0.f ? v : 0.f);
            }
        }
    } else {
#pragma unroll
        for (int nb = 0; nb < 4; ++nb) {
            const int col = nb * 16 + n0;
            const float b = bias[col];
#pragma unroll
            for (int r = 0; r < 8; ++r) {
                const int row = rbase + r;
                if (row < rows_valid) {
                    float v = acc[nb][r] + b;
                    out[row * ldo + col] = f2bf(v > 0.f ? v : 0.f);
                }
            }
        }
    }
}

// ---------------------------------------------------------------------------
// Weight packing: fp32 row-major [Ktot,64] -> bf16 fragment order.
// dst[idx], idx = ((kc*4+nb)*32 + lane)*16 + i  maps to
//   K = kc*32 + (lane&16 ? 8:0) + (i<8 ? i : 8+i) ; col = nb*16 + (lane&15)
// ---------------------------------------------------------------------------
__global__ void gnn_pack_w(const float* __restrict__ src, unsigned short* __restrict__ dst, int Ktot) {
    const int total = Ktot * 64;
    for (int idx = blockIdx.x * blockDim.x + threadIdx.x; idx < total; idx += gridDim.x * blockDim.x) {
        const int i  = idx & 15;
        const int l  = (idx >> 4) & 31;
        const int c  = idx >> 9;          // chunk-block index = kc*4 + nb
        const int nb = c & 3;
        const int kc = c >> 2;
        const int K   = kc * 32 + ((l & 16) ? 8 : 0) + (i < 8 ? i : 8 + i);
        const int col = nb * 16 + (l & 15);
        dst[idx] = f2bf(src[K * 64 + col]);
    }
}

__global__ void gnn_zero_f32(float* __restrict__ p, int n) {
    for (int i = blockIdx.x * blockDim.x + threadIdx.x; i < n; i += gridDim.x * blockDim.x)
        p[i] = 0.f;
}

// ---------------------------------------------------------------------------
// Node encoder: x[N,2] -> relu(xW1+b1) -> relu(hW2+b2) -> x0, xc (bf16)
// ---------------------------------------------------------------------------
__global__ void gnn_node_encoder(const float* __restrict__ x,
                                 const float* __restrict__ W1, const float* __restrict__ b1,
                                 const unsigned short* __restrict__ W2p, const float* __restrict__ b2,
                                 int N, unsigned short* __restrict__ x0, unsigned short* __restrict__ xc) {
    __shared__ __align__(16) unsigned short sH[64 * 64];
    const int tile0 = blockIdx.x * 64;
    for (int idx = threadIdx.x; idx < 64 * 64; idx += blockDim.x) {
        const int r = idx >> 6, j = idx & 63;
        const int row = tile0 + r;
        const int rr  = row < N ? row : 0;
        float h = x[rr * 2 + 0] * W1[j] + x[rr * 2 + 1] * W1[64 + j] + b1[j];
        h = h > 0.f ? h : 0.f;
        sH[r * 64 + j] = f2bf(row < N ? h : 0.f);
    }
    __syncthreads();
    const int w = threadIdx.x >> 5;
    v8f acc[4] = {};
    wmma_tile<64, 64>(sH + w * 16 * 64, W2p, acc);
    const int row0 = tile0 + w * 16;
    epilogue_relu(acc, b2, x0 + (size_t)row0 * 64, 64, N - row0);
    epilogue_relu(acc, b2, xc + (size_t)row0 * 64, 64, N - row0);
}

// ---------------------------------------------------------------------------
// Edge encoder: ef[E,3] -> ... -> e0, ec (bf16)
// ---------------------------------------------------------------------------
__global__ void gnn_edge_encoder(const float* __restrict__ ef,
                                 const float* __restrict__ W1, const float* __restrict__ b1,
                                 const unsigned short* __restrict__ W2p, const float* __restrict__ b2,
                                 int E, unsigned short* __restrict__ e0, unsigned short* __restrict__ ec) {
    __shared__ __align__(16) unsigned short sH[64 * 64];
    const int tile0 = blockIdx.x * 64;
    for (int idx = threadIdx.x; idx < 64 * 64; idx += blockDim.x) {
        const int r = idx >> 6, j = idx & 63;
        const int row = tile0 + r;
        const int rr  = row < E ? row : 0;
        float h = ef[rr * 3 + 0] * W1[j] + ef[rr * 3 + 1] * W1[64 + j] +
                  ef[rr * 3 + 2] * W1[128 + j] + b1[j];
        h = h > 0.f ? h : 0.f;
        sH[r * 64 + j] = f2bf(row < E ? h : 0.f);
    }
    __syncthreads();
    const int w = threadIdx.x >> 5;
    v8f acc[4] = {};
    wmma_tile<64, 64>(sH + w * 16 * 64, W2p, acc);
    const int row0 = tile0 + w * 16;
    epilogue_relu(acc, b2, e0 + (size_t)row0 * 64, 64, E - row0);
    epilogue_relu(acc, b2, ec + (size_t)row0 * 64, 64, E - row0);
}

// ---------------------------------------------------------------------------
// Scatter-add: aggr[dst] += concat(ec, e0)[edge]  (4 cols/thread, f32 atomics)
// ---------------------------------------------------------------------------
__global__ void gnn_scatter_add(const unsigned short* __restrict__ ec,
                                const unsigned short* __restrict__ e0,
                                const int* __restrict__ ei, int E,
                                float* __restrict__ aggr) {
    const int total  = E * 32;
    const int stride = gridDim.x * blockDim.x;
    for (int idx = blockIdx.x * blockDim.x + threadIdx.x; idx < total; idx += stride) {
        const int e  = idx >> 5;
        const int c4 = (idx & 31) << 2;          // 0,4,...,124
        const int dst = ei[E + e];
        const unsigned short* sp = (c4 < 64) ? (ec + (size_t)e * 64 + c4)
                                             : (e0 + (size_t)e * 64 + (c4 - 64));
        const uint2 u = *(const uint2*)sp;       // 4 bf16
        float* dp = aggr + (size_t)dst * 128 + c4;
        atomicAdd(dp + 0, bf2f((unsigned short)(u.x & 0xFFFFu)));
        atomicAdd(dp + 1, bf2f((unsigned short)(u.x >> 16)));
        atomicAdd(dp + 2, bf2f((unsigned short)(u.y & 0xFFFFu)));
        atomicAdd(dp + 3, bf2f((unsigned short)(u.y >> 16)));
    }
}

// ---------------------------------------------------------------------------
// Edge update: ext = [ec,e0, xc[src],x0[src], xc[dst],x0[dst]]  (K=384)
//              ec' = relu(relu(ext W1 + b1) W2 + b2)
// ---------------------------------------------------------------------------
__global__ void gnn_edge_update(const unsigned short* __restrict__ ec,
                                const unsigned short* __restrict__ e0,
                                const unsigned short* __restrict__ xc,
                                const unsigned short* __restrict__ x0,
                                const int* __restrict__ ei, int E,
                                const unsigned short* __restrict__ W1p, const float* __restrict__ b1,
                                const unsigned short* __restrict__ W2p, const float* __restrict__ b2,
                                unsigned short* __restrict__ ecOut) {
    __shared__ __align__(16) unsigned short sA[64 * 384];   // 48 KB
    __shared__ __align__(16) unsigned short sH[64 * 64];    //  8 KB
    const int tile0 = blockIdx.x * 64;

    const uint4* ecq = (const uint4*)ec;   // 8 quads per 64-bf16 row
    const uint4* e0q = (const uint4*)e0;
    const uint4* xcq = (const uint4*)xc;
    const uint4* x0q = (const uint4*)x0;
    uint4* sAq = (uint4*)sA;               // row stride = 48 quads

    for (int idx = threadIdx.x; idx < 64 * 48; idx += blockDim.x) {
        const int r = idx / 48, d = idx % 48;
        const int e  = tile0 + r;
        const bool ok = e < E;
        const int ee = ok ? e : 0;
        const int seg = d >> 3, off = d & 7;
        const int src = ei[ee];
        const int dst = ei[E + ee];
        uint4 val;
        switch (seg) {
            case 0:  val = ecq[(size_t)ee * 8 + off];  break;
            case 1:  val = e0q[(size_t)ee * 8 + off];  break;
            case 2:  val = xcq[(size_t)src * 8 + off]; break;
            case 3:  val = x0q[(size_t)src * 8 + off]; break;
            case 4:  val = xcq[(size_t)dst * 8 + off]; break;
            default: val = x0q[(size_t)dst * 8 + off]; break;
        }
        if (!ok) { val.x = 0; val.y = 0; val.z = 0; val.w = 0; }
        sAq[r * 48 + d] = val;
    }
    __syncthreads();

    const int w = threadIdx.x >> 5;
    v8f acc[4] = {};
    wmma_tile<384, 384>(sA + w * 16 * 384, W1p, acc);       // 12 chunks x 4 wmma
    unsigned short* H = sH + w * 16 * 64;
    epilogue_relu(acc, b1, H, 64, 16);                      // wave-local hidden

    v8f acc2[4] = {};
    wmma_tile<64, 64>(H, W2p, acc2);
    const int row0 = tile0 + w * 16;
    epilogue_relu(acc2, b2, ecOut + (size_t)row0 * 64, 64, E - row0);
}

// ---------------------------------------------------------------------------
// Node update: in = [xc, x0, aggr(128)]  (K=256)
//              xc' = relu(relu(in W1 + b1) W2 + b2)
// ---------------------------------------------------------------------------
__global__ void gnn_node_update(const unsigned short* __restrict__ xc,
                                const unsigned short* __restrict__ x0,
                                const float* __restrict__ aggr, int N,
                                const unsigned short* __restrict__ W1p, const float* __restrict__ b1,
                                const unsigned short* __restrict__ W2p, const float* __restrict__ b2,
                                unsigned short* __restrict__ xcOut) {
    __shared__ __align__(16) unsigned short sA[64 * 256];   // 32 KB
    __shared__ __align__(16) unsigned short sH[64 * 64];
    const int tile0 = blockIdx.x * 64;

    const uint4* xcq = (const uint4*)xc;
    const uint4* x0q = (const uint4*)x0;
    uint4* sAq = (uint4*)sA;                                // row stride = 32 quads
    for (int idx = threadIdx.x; idx < 64 * 16; idx += blockDim.x) {   // xc | x0 halves
        const int r = idx >> 4, d = idx & 15;
        const int node = tile0 + r;
        const bool ok = node < N;
        const int nn = ok ? node : 0;
        uint4 val = (d < 8) ? xcq[(size_t)nn * 8 + d] : x0q[(size_t)nn * 8 + (d - 8)];
        if (!ok) { val.x = 0; val.y = 0; val.z = 0; val.w = 0; }
        sAq[r * 32 + d] = val;
    }
    const float4* agq = (const float4*)aggr;                // 32 float4 per row
    uint2* sA2 = (uint2*)sA;                                // row stride = 64 uint2
    for (int idx = threadIdx.x; idx < 64 * 32; idx += blockDim.x) {   // aggr f32->bf16
        const int r = idx >> 5, q = idx & 31;
        const int node = tile0 + r;
        const bool ok = node < N;
        const int nn = ok ? node : 0;
        float4 v = agq[(size_t)nn * 32 + q];
        if (!ok) { v.x = v.y = v.z = v.w = 0.f; }
        uint2 o;
        o.x = (unsigned int)f2bf(v.x) | ((unsigned int)f2bf(v.y) << 16);
        o.y = (unsigned int)f2bf(v.z) | ((unsigned int)f2bf(v.w) << 16);
        sA2[r * 64 + 32 + q] = o;
    }
    __syncthreads();

    const int w = threadIdx.x >> 5;
    v8f acc[4] = {};
    wmma_tile<256, 256>(sA + w * 16 * 256, W1p, acc);       // 8 chunks x 4 wmma
    unsigned short* H = sH + w * 16 * 64;
    epilogue_relu(acc, b1, H, 64, 16);

    v8f acc2[4] = {};
    wmma_tile<64, 64>(H, W2p, acc2);
    const int row0 = tile0 + w * 16;
    epilogue_relu(acc2, b2, xcOut + (size_t)row0 * 64, 64, N - row0);
}

// ---------------------------------------------------------------------------
// Decoder: out = relu(xc W1 + b1) @ W2 + b2   (64 -> 64 -> 1)
// ---------------------------------------------------------------------------
__global__ void gnn_decoder(const unsigned short* __restrict__ xc,
                            const unsigned short* __restrict__ W1p, const float* __restrict__ b1,
                            const float* __restrict__ W2, const float* __restrict__ b2,
                            int N, float* __restrict__ out) {
    __shared__ __align__(16) unsigned short sA[64 * 64];
    __shared__ __align__(16) unsigned short sH[64 * 64];
    const int tile0 = blockIdx.x * 64;

    const uint4* xcq = (const uint4*)xc;
    uint4* sAq = (uint4*)sA;
    for (int idx = threadIdx.x; idx < 64 * 8; idx += blockDim.x) {
        const int r = idx >> 3, d = idx & 7;
        const int node = tile0 + r;
        const bool ok = node < N;
        uint4 val;
        if (ok) val = xcq[(size_t)node * 8 + d];
        else    { val.x = 0; val.y = 0; val.z = 0; val.w = 0; }
        sAq[r * 8 + d] = val;
    }
    __syncthreads();

    const int w = threadIdx.x >> 5;
    v8f acc[4] = {};
    wmma_tile<64, 64>(sA + w * 16 * 64, W1p, acc);
    epilogue_relu(acc, b1, sH + w * 16 * 64, 64, 16);
    __syncthreads();

    for (int r = threadIdx.x; r < 64; r += blockDim.x) {
        const int node = tile0 + r;
        if (node < N) {
            float s = b2[0];
#pragma unroll
            for (int j = 0; j < 64; ++j)
                s += bf2f(sH[r * 64 + j]) * W2[j];
            out[node] = s;      // no final ReLU
        }
    }
}

// ---------------------------------------------------------------------------
// Host launcher
// ---------------------------------------------------------------------------
extern "C" void kernel_launch(void* const* d_in, const int* in_sizes, int n_in,
                              void* d_out, int out_size, void* d_ws, size_t ws_size,
                              hipStream_t stream) {
    const float* x       = (const float*)d_in[0];
    const float* ef      = (const float*)d_in[1];
    const float* ne_W1   = (const float*)d_in[2];
    const float* ne_b1   = (const float*)d_in[3];
    const float* ne_W2   = (const float*)d_in[4];
    const float* ne_b2   = (const float*)d_in[5];
    const float* ee_W1   = (const float*)d_in[6];
    const float* ee_b1   = (const float*)d_in[7];
    const float* ee_W2   = (const float*)d_in[8];
    const float* ee_b2   = (const float*)d_in[9];
    const float* eu_W1   = (const float*)d_in[10];
    const float* eu_b1   = (const float*)d_in[11];
    const float* eu_W2   = (const float*)d_in[12];
    const float* eu_b2   = (const float*)d_in[13];
    const float* nu_W1   = (const float*)d_in[14];
    const float* nu_b1   = (const float*)d_in[15];
    const float* nu_W2   = (const float*)d_in[16];
    const float* nu_b2   = (const float*)d_in[17];
    const float* dec_W1  = (const float*)d_in[18];
    const float* dec_b1  = (const float*)d_in[19];
    const float* dec_W2  = (const float*)d_in[20];
    const float* dec_b2  = (const float*)d_in[21];
    const int*   ei      = (const int*)d_in[22];
    float* out = (float*)d_out;

    const int N = in_sizes[0] / 2;
    const int E = in_sizes[1] / 3;
    const int L = in_sizes[10] / (6 * 64 * 64);

    // ---- carve workspace ----
    size_t off = 0;
    auto carve = [&](size_t bytes) -> void* {
        void* p = (char*)d_ws + off;
        off += (bytes + 255) & ~(size_t)255;
        return p;
    };
    unsigned short* x0   = (unsigned short*)carve((size_t)N * 64 * 2);
    unsigned short* xcA  = (unsigned short*)carve((size_t)N * 64 * 2);
    unsigned short* xcB  = (unsigned short*)carve((size_t)N * 64 * 2);
    unsigned short* e0   = (unsigned short*)carve((size_t)E * 64 * 2);
    unsigned short* ecA  = (unsigned short*)carve((size_t)E * 64 * 2);
    unsigned short* ecB  = (unsigned short*)carve((size_t)E * 64 * 2);
    float*          aggr = (float*)carve((size_t)N * 128 * 4);
    unsigned short* wNE2 = (unsigned short*)carve(4096 * 2);
    unsigned short* wEE2 = (unsigned short*)carve(4096 * 2);
    unsigned short* wDEC = (unsigned short*)carve(4096 * 2);
    unsigned short* wEU1 = (unsigned short*)carve((size_t)L * 384 * 64 * 2);
    unsigned short* wEU2 = (unsigned short*)carve((size_t)L * 64 * 64 * 2);
    unsigned short* wNU1 = (unsigned short*)carve((size_t)L * 256 * 64 * 2);
    unsigned short* wNU2 = (unsigned short*)carve((size_t)L * 64 * 64 * 2);

    // ---- weight packing into fragment order (cheap; rerun each call) ----
    auto pack = [&](const float* s, unsigned short* d, int Ktot) {
        gnn_pack_w<<<(Ktot * 64 + 255) / 256, 256, 0, stream>>>(s, d, Ktot);
    };
    pack(ne_W2, wNE2, 64);
    pack(ee_W2, wEE2, 64);
    pack(dec_W1, wDEC, 64);
    pack(eu_W1, wEU1, L * 384);   // chunk layout is K-periodic; layers concat cleanly
    pack(eu_W2, wEU2, L * 64);
    pack(nu_W1, wNU1, L * 256);
    pack(nu_W2, wNU2, L * 64);

    const int nodeTiles = (N + 63) / 64;
    const int edgeTiles = (E + 63) / 64;

    // ---- encoders ----
    gnn_node_encoder<<<nodeTiles, 128, 0, stream>>>(x, ne_W1, ne_b1, wNE2, ne_b2, N, x0, xcA);
    gnn_edge_encoder<<<edgeTiles, 128, 0, stream>>>(ef, ee_W1, ee_b1, wEE2, ee_b2, E, e0, ecA);

    // ---- message-passing layers ----
    unsigned short* xcCur = xcA; unsigned short* xcNxt = xcB;
    unsigned short* ecCur = ecA; unsigned short* ecNxt = ecB;
    for (int l = 0; l < L; ++l) {
        gnn_zero_f32<<<(N * 128 + 255) / 256, 256, 0, stream>>>(aggr, N * 128);
        gnn_scatter_add<<<4096, 256, 0, stream>>>(ecCur, e0, ei, E, aggr);
        gnn_edge_update<<<edgeTiles, 128, 0, stream>>>(
            ecCur, e0, xcCur, x0, ei, E,
            wEU1 + (size_t)l * 384 * 64, eu_b1 + l * 64,
            wEU2 + (size_t)l * 64 * 64,  eu_b2 + l * 64, ecNxt);
        gnn_node_update<<<nodeTiles, 128, 0, stream>>>(
            xcCur, x0, aggr, N,
            wNU1 + (size_t)l * 256 * 64, nu_b1 + l * 64,
            wNU2 + (size_t)l * 64 * 64,  nu_b2 + l * 64, xcNxt);
        unsigned short* t;
        t = xcCur; xcCur = xcNxt; xcNxt = t;
        t = ecCur; ecCur = ecNxt; ecNxt = t;
    }

    // ---- decoder ----
    gnn_decoder<<<nodeTiles, 128, 0, stream>>>(xcCur, wDEC, dec_b1, dec_W2, dec_b2, N, out);
}
